// NTXentLoss_5746666242426
// MI455X (gfx1250) — matrix operand order repaired and checked by reading, not compiled, not run on hardware
//
#include <hip/hip_runtime.h>

typedef __attribute__((ext_vector_type(16))) _Float16 v16h;
typedef __attribute__((ext_vector_type(8)))  _Float16 v8h;
typedef __attribute__((ext_vector_type(8)))  float    v8f;

#define N_ROWS 8192
#define DIMK   256
#define TILES  (N_ROWS / 16)   // 512 column tiles of 16

// ---------------------------------------------------------------------------
__global__ void zero_out_kernel(float* out) { out[0] = 0.0f; }

// ---------------------------------------------------------------------------
// Kernel 1: interleave rows (even -> zjs, odd -> zis), L2-normalize with the
// reference's eps clamp, emit f16 zn[8192][256] into workspace.
// ---------------------------------------------------------------------------
__global__ __launch_bounds__(256)
void normalize_kernel(const float* __restrict__ zis,
                      const float* __restrict__ zjs,
                      _Float16* __restrict__ zn) {
  const int lane = threadIdx.x & 31;
  const int wave = threadIdx.x >> 5;
  const int row  = blockIdx.x * 8 + wave;

  const float* src = (row & 1) ? (zis + (size_t)(row >> 1) * DIMK)
                               : (zjs + (size_t)(row >> 1) * DIMK);
  const float4* p = (const float4*)(src + lane * 8);
  float4 a = p[0], b = p[1];

  float ss = a.x * a.x + a.y * a.y + a.z * a.z + a.w * a.w
           + b.x * b.x + b.y * b.y + b.z * b.z + b.w * b.w;
#pragma unroll
  for (int m = 16; m >= 1; m >>= 1) ss += __shfl_xor(ss, m, 32);

  const float inv = 1.0f / fmaxf(sqrtf(ss), 1e-8f);

  v8h h;
  h[0] = (_Float16)(a.x * inv); h[1] = (_Float16)(a.y * inv);
  h[2] = (_Float16)(a.z * inv); h[3] = (_Float16)(a.w * inv);
  h[4] = (_Float16)(b.x * inv); h[5] = (_Float16)(b.y * inv);
  h[6] = (_Float16)(b.z * inv); h[7] = (_Float16)(b.w * inv);
  *(v8h*)(zn + (size_t)row * DIMK + lane * 8) = h;
}

// ---------------------------------------------------------------------------
// Kernel 2: 256 workgroups; each owns 32 rows (2 A-tiles per wave, resident in
// registers). 8 waves split the 512 column tiles. Per tile: one clause of 16
// global_load_b128 (B) into distinct regs, pipelined waits + 16
// v_wmma_f32_16x16x32_f16, WGP-scope prefetch of the next tile at a constant
// immediate offset (no extra address math), then branch-free exp accumulation
// (diagonal tile handled behind a wave-uniform branch).
// ---------------------------------------------------------------------------
__global__ __launch_bounds__(256)
void ntxent_kernel(const _Float16* __restrict__ zn, float* __restrict__ out) {
  __shared__ float rowsum[32];
  __shared__ float tgt[32];

  const int tid = threadIdx.x;
  if (tid < 32) { rowsum[tid] = 0.0f; tgt[tid] = 0.0f; }
  __syncthreads();

  const int lane    = tid & 31;
  const int wave    = tid >> 5;
  const int halfsel = lane >> 4;        // 0: lanes 0-15, 1: lanes 16-31
  const int l15     = lane & 15;
  const int Mbase   = halfsel ? 8 : 0;  // C/D: vgpr v <-> row M = v + 8*half
  const int it0     = blockIdx.x * 2;   // first of two row tiles
  const int it1     = it0 + 1;

  // ---- A strips in ISA 16-bit A layout ------------------------------------
  // lanes 0-15  (M=lane): halves = K{k0..k0+7}  then K{k0+16..k0+23}
  // lanes 16-31 (M=l15) : halves = K{k0+8..+15} then K{k0+24..+31}
  v16h A0[8], A1[8];
  {
    const _Float16* ar0 = zn + (size_t)(it0 * 16 + l15) * DIMK + halfsel * 8;
    const _Float16* ar1 = ar0 + 16 * DIMK;
#pragma unroll
    for (int kc = 0; kc < 8; ++kc) {
      v8h s0 = *(const v8h*)(ar0 + kc * 32);
      v8h s1 = *(const v8h*)(ar0 + kc * 32 + 16);
      A0[kc] = __builtin_shufflevector(s0, s1, 0, 1, 2, 3, 4, 5, 6, 7,
                                               8, 9, 10, 11, 12, 13, 14, 15);
      v8h t0 = *(const v8h*)(ar1 + kc * 32);
      v8h t1 = *(const v8h*)(ar1 + kc * 32 + 16);
      A1[kc] = __builtin_shufflevector(t0, t1, 0, 1, 2, 3, 4, 5, 6, 7,
                                               8, 9, 10, 11, 12, 13, 14, 15);
    }
  }

  float acc0[8], acc1[8];
#pragma unroll
  for (int v = 0; v < 8; ++v) { acc0[v] = 0.0f; acc1[v] = 0.0f; }

  for (int jt = wave; jt < TILES; jt += 8) {
    // ---- B tile, ISA 16-bit B layout (N = l15, K same halves as A) -----
    const _Float16* brow = zn + (size_t)(jt * 16 + l15) * DIMK + halfsel * 8;
    v16h B[8];
#pragma unroll
    for (int kc = 0; kc < 8; ++kc) {
      v8h s0 = *(const v8h*)(brow + kc * 32);
      v8h s1 = *(const v8h*)(brow + kc * 32 + 16);
      B[kc] = __builtin_shufflevector(s0, s1, 0, 1, 2, 3, 4, 5, 6, 7,
                                              8, 9, 10, 11, 12, 13, 14, 15);
    }

    // ---- WGP-scope prefetch of next tile: same base reg, immediate offset.
    // Next tile for this wave is +8 tiles = +65536 B; cover all four 128-B
    // lines of the 512-B row. No extra VGPRs, no address math.
    if (jt + 8 < TILES) {
      const char* nb = (const char*)brow;
      asm volatile("global_prefetch_b8 %0, off offset:65536" :: "v"(nb));
      asm volatile("global_prefetch_b8 %0, off offset:65664" :: "v"(nb));
      asm volatile("global_prefetch_b8 %0, off offset:65792" :: "v"(nb));
      asm volatile("global_prefetch_b8 %0, off offset:65920" :: "v"(nb));
    }

    v8f c0 = {}, c1 = {};
#pragma unroll
    for (int kc = 0; kc < 8; ++kc) {
      c0 = __builtin_amdgcn_wmma_f32_16x16x32_f16(false, A0[kc], false, B[kc],
                                                  (short)0, c0, false, false);
      c1 = __builtin_amdgcn_wmma_f32_16x16x32_f16(false, A1[kc], false, B[kc],
                                                  (short)0, c1, false, false);
    }

    // ---- epilogue: wave-uniform diagonal branch ---------------------------
    if (jt == it0) {            // diagonal tile for c0 (taken once, by wave it0 % 8 only)
#pragma unroll
      for (int v = 0; v < 8; ++v) {
        const int m = Mbase + v;
        const float s = 2.0f * c0[v];
        if (l15 == (m ^ 1)) tgt[m] = s;                  // target logit
        acc0[v] += (l15 == m) ? 0.0f : __expf(s);        // mask diagonal
      }
    } else {
#pragma unroll
      for (int v = 0; v < 8; ++v) acc0[v] += __expf(2.0f * c0[v]);
    }
    if (jt == it1) {            // diagonal tile for c1
#pragma unroll
      for (int v = 0; v < 8; ++v) {
        const int m = Mbase + v;
        const float s = 2.0f * c1[v];
        if (l15 == (m ^ 1)) tgt[16 + m] = s;
        acc1[v] += (l15 == m) ? 0.0f : __expf(s);
      }
    } else {
#pragma unroll
      for (int v = 0; v < 8; ++v) acc1[v] += __expf(2.0f * c1[v]);
    }
  }

#pragma unroll
  for (int v = 0; v < 8; ++v) {
    atomicAdd(&rowsum[Mbase + v], acc0[v]);
    atomicAdd(&rowsum[16 + Mbase + v], acc1[v]);
  }
  __syncthreads();

  if (tid < 32) {
    float li = __logf(rowsum[tid]) - tgt[tid];
#pragma unroll
    for (int m = 16; m >= 1; m >>= 1) li += __shfl_xor(li, m, 32);
    if (tid == 0) atomicAdd(out, li * (1.0f / (float)N_ROWS));
  }
}

// ---------------------------------------------------------------------------
extern "C" void kernel_launch(void* const* d_in, const int* in_sizes, int n_in,
                              void* d_out, int out_size, void* d_ws, size_t ws_size,
                              hipStream_t stream) {
  const float* zis = (const float*)d_in[0];
  const float* zjs = (const float*)d_in[1];
  float* out = (float*)d_out;
  _Float16* zn = (_Float16*)d_ws;  // 8192 * 256 * 2B = 4 MiB

  zero_out_kernel<<<1, 1, 0, stream>>>(out);
  normalize_kernel<<<N_ROWS / 8, 256, 0, stream>>>(zis, zjs, zn);
  ntxent_kernel<<<N_ROWS / 32, 256, 0, stream>>>(zn, out);
}
